// TemplatePointwiseAttention_70214125355311
// MI455X (gfx1250) — matrix operand order repaired and checked by reading, not compiled, not run on hardware
//
#include <hip/hip_runtime.h>
#include <hip/hip_bf16.h>

typedef __attribute__((ext_vector_type(16))) _Float16 v16h;
typedef __attribute__((ext_vector_type(8)))  _Float16 v8h;
typedef __attribute__((ext_vector_type(4)))  _Float16 v4h;
typedef __attribute__((ext_vector_type(8)))  float    v8f;
typedef __attribute__((ext_vector_type(4)))  float    v4f;
typedef __attribute__((ext_vector_type(4)))  unsigned int v4u;
typedef __attribute__((ext_vector_type(8)))  int      v8i;
typedef __attribute__((ext_vector_type(4)))  int      v4i;

#define RR    384
#define NPAIR (RR * RR)   // 147456
#define TM    32          // pairs per block
#define NTPL  4           // templates
#define DT    64          // t feature dim
#define DZ    128         // z feature dim
#define NH    4           // heads
#define HD    16          // per-head dim
#define HID   64          // NH*HD

// prepped-weight layout inside d_ws (f16 halves):
//   [0      .. 8191 ]  wqT [n:64][k:128]
//   [8192   .. 12287]  wkT [n:64][k:64]
//   [12288  .. 16383]  wvT [n:64][k:64]
//   [16384  .. 24575]  woT [n:128][k:64]
#define WPRE_TOTAL 24576

// ---- WMMA fragment helpers (ISA 7.12.2 layouts, wave32) -------------------

// A-matrix 16x32 f16: lane l -> row l&15; halves 0..7 = K kb..kb+7,
// halves 8..15 = K kb+16..kb+23, kb = (l>>4)*8.
__device__ __forceinline__ v16h load_a_frag(const _Float16* base, int stride) {
  const int lane = threadIdx.x & 31;
  const int row  = lane & 15;
  const int kb   = (lane >> 4) << 3;
  const _Float16* p = base + row * stride + kb;
  v8h lo = *(const v8h*)(p);
  v8h hi = *(const v8h*)(p + 16);
  v16h r;
#pragma unroll
  for (int i = 0; i < 8; ++i) { r[i] = lo[i]; r[i + 8] = hi[i]; }
  return r;
}

// B-matrix 32x16 f16 from transposed [N][K] storage: lane l -> col l&15;
// halves 0..7 = K kb..kb+7, halves 8..15 = K kb+16..kb+23, kb = (l>>4)*8.
__device__ __forceinline__ v16h load_b_frag(const _Float16* base, int stride) {
  const int lane = threadIdx.x & 31;
  const int n    = lane & 15;
  const int kb   = (lane >> 4) << 3;
  const _Float16* p = base + n * stride + kb;
  v8h lo = *(const v8h*)(p);
  v8h hi = *(const v8h*)(p + 16);
  v16h r;
#pragma unroll
  for (int i = 0; i < 8; ++i) { r[i] = lo[i]; r[i + 8] = hi[i]; }
  return r;
}

__device__ __forceinline__ v8f wmma32(v16h a, v16h b, v8f c) {
  return __builtin_amdgcn_wmma_f32_16x16x32_f16(false, a, false, b, (short)0,
                                                c, false, false);
}

// ---- One-shot weight transpose + f32->f16 convert into d_ws ---------------
__global__ void __launch_bounds__(256)
prep_weights(const float* __restrict__ wq, const float* __restrict__ wk,
             const float* __restrict__ wv, const float* __restrict__ wo,
             _Float16* __restrict__ ws)
{
  const int i = blockIdx.x * 256 + threadIdx.x;   // grid covers WPRE_TOTAL
  if (i < 8192) {                                  // wq [128][64] -> wqT
    const int k = i >> 6, n = i & 63;
    ws[n * DZ + k] = (_Float16)wq[i];
  } else if (i < 12288) {                          // wk [64][64] -> wkT
    const int j = i - 8192, k = j >> 6, n = j & 63;
    ws[8192 + n * DT + k] = (_Float16)wk[j];
  } else if (i < 16384) {                          // wv [64][64] -> wvT
    const int j = i - 12288, k = j >> 6, n = j & 63;
    ws[12288 + n * DT + k] = (_Float16)wv[j];
  } else if (i < WPRE_TOTAL) {                     // wo [64][128] -> woT
    const int j = i - 16384, k = j >> 7, n = j & 127;
    ws[16384 + n * HID + k] = (_Float16)wo[j];
  }
}

// ---------------------------------------------------------------------------

__global__ void __launch_bounds__(256, 2)
tpa_kernel(const float* __restrict__ gT,  const float* __restrict__ gZ,
           const float* __restrict__ gMask,
           const _Float16* __restrict__ gWpre,
           const float* __restrict__ gBo, float* __restrict__ gOut)
{
  __shared__ __align__(16) _Float16 sZ[TM * DZ];          //  8 KB
  __shared__ __align__(16) _Float16 sT[TM * NTPL * DT];   // 16 KB  [m][tt][k]
  __shared__ __align__(16) _Float16 sW[WPRE_TOTAL];       // 48 KB  packed wts
  __shared__ __align__(16) _Float16 sQ[TM * HID];         //  4 KB
  __shared__ __align__(16) _Float16 sK[TM * NTPL * DT];   // 16 KB
  __shared__ __align__(16) _Float16 sV[TM * NTPL * DT];   // 16 KB
  __shared__ __align__(16) _Float16 sO[TM * HID];         //  4 KB
  __shared__ float sBo[DZ];
  __shared__ float sBias[NTPL];

  const _Float16* sWqT = sW;
  const _Float16* sWkT = sW + 8192;
  const _Float16* sWvT = sW + 12288;
  const _Float16* sWoT = sW + 16384;

  const int tid  = threadIdx.x;
  const int wave = tid >> 5;
  const int lane = tid & 31;
  const int p0   = blockIdx.x * TM;

  // ---- Phase 0a: kick off TDM DMA of prepped f16 weights -> LDS -----------
  // One wave issues a single tensor_load_to_lds covering all 24576 halves
  // (1-row 2D tensor, tile == tensor, data_size = 2 bytes). Tracked by
  // TENSORcnt; the DMA runs while all waves stage z/t below.
  if (wave == 0) {
    const unsigned long long ga = (unsigned long long)(const void*)gWpre;
    v4u g0;
    g0[0] = 1u;                                   // count=1, user descriptor
    g0[1] = (unsigned int)(unsigned long long)(const void*)sW;  // lds_addr
    g0[2] = (unsigned int)(ga & 0xFFFFFFFFull);   // global_addr[31:0]
    g0[3] = ((unsigned int)(ga >> 32) & 0x01FFFFFFu) | (2u << 30); // type=2
    v8i g1;
    g1[0] = (1 << 16);                 // wg_mask=0, data_size=1 (2 bytes)
    g1[1] = (int)((WPRE_TOTAL & 0xFFFF) << 16);   // tensor_dim0[15:0]
    g1[2] = (int)((WPRE_TOTAL >> 16) | (1 << 16)); // dim0[31:16], tensor_dim1=1
    g1[3] = (int)((WPRE_TOTAL & 0xFFFF) << 16);   // tile_dim0 = 24576
    g1[4] = 1;                          // tile_dim1=1, tile_dim2=0
    g1[5] = WPRE_TOTAL;                 // tensor_dim0_stride[31:0]
    g1[6] = 0;                          // stride0[47:32], dim1_stride[15:0]
    g1[7] = 0;
    v4i g2 = {0, 0, 0, 0};              // group 2 unused (2-D tensor)
    v4i g3 = {0, 0, 0, 0};              // group 3 unused
    v8i gx = {0, 0, 0, 0, 0, 0, 0, 0};  // extra group (clang-23 6-arg form)
    __builtin_amdgcn_tensor_load_to_lds(g0, g1, g2, g3, gx, 0);
  }

  // ---- Phase 0b: stage z/t tiles into LDS as f16 (VMEM + cvt) -------------
  // z tile: TM x DZ floats -> f16
#pragma unroll
  for (int idx = tid; idx < TM * DZ / 4; idx += 256) {
    v4f f = *(const v4f*)(gZ + (size_t)p0 * DZ + idx * 4);
    v4h h;
#pragma unroll
    for (int j = 0; j < 4; ++j) h[j] = (_Float16)f[j];
    *(v4h*)(sZ + idx * 4) = h;
  }
  // t tile: NTPL planes of TM x DT floats -> sT[m][tt][k]
#pragma unroll
  for (int idx = tid; idx < NTPL * TM * DT / 4; idx += 256) {
    int tt  = idx / (TM * DT / 4);
    int rem = idx % (TM * DT / 4);
    int row = rem / (DT / 4);
    int c4  = rem % (DT / 4);
    v4f f = *(const v4f*)(gT + ((size_t)tt * NPAIR + p0 + row) * DT + c4 * 4);
    v4h h;
#pragma unroll
    for (int j = 0; j < 4; ++j) h[j] = (_Float16)f[j];
    *(v4h*)(sT + (row * NTPL + tt) * DT + c4 * 4) = h;
  }
  if (tid < DZ)   sBo[tid]   = gBo[tid];
  if (tid < NTPL) sBias[tid] = (gMask[tid] > 0.0f) ? 0.0f : -1e9f;
  if (wave == 0) __builtin_amdgcn_s_wait_tensorcnt(0);  // weight DMA done
  __syncthreads();

  // ---- Phase 1: Q = (z @ wq) * 1/sqrt(HD)   [TM x HID], K=128 ------------
  {
    const int mt = wave >> 2, nt = wave & 3;   // exactly 8 jobs = 8 waves
    v8f acc = {};
#pragma unroll
    for (int k0 = 0; k0 < DZ; k0 += 32) {
      v16h a = load_a_frag(sZ + (mt * 16) * DZ + k0, DZ);
      v16h b = load_b_frag(sWqT + (nt * 16) * DZ + k0, DZ);
      acc = wmma32(a, b, acc);
    }
    const int col = lane & 15, rb = (lane >> 4) << 3;
#pragma unroll
    for (int i = 0; i < 8; ++i)
      sQ[(mt * 16 + rb + i) * HID + nt * 16 + col] = (_Float16)(acc[i] * 0.25f);
  }
  __syncthreads();

  // ---- Phase 2: K = t @ wk, V = t @ wv per template, K=64 -----------------
#pragma unroll
  for (int job = wave; job < 64; job += 8) {
    const int kv  = job >> 5;
    const int rem = job & 31;
    const int tt  = rem >> 3;
    const int mt  = (rem >> 2) & 1;
    const int nt  = rem & 3;
    const _Float16* Bw  = kv ? sWvT : sWkT;
    _Float16*       Dst = kv ? (_Float16*)sV : (_Float16*)sK;
    v8f acc = {};
#pragma unroll
    for (int k0 = 0; k0 < DT; k0 += 32) {
      v16h a = load_a_frag(sT + (mt * 16) * (NTPL * DT) + tt * DT + k0,
                           NTPL * DT);
      v16h b = load_b_frag(Bw + (nt * 16) * DT + k0, DT);
      acc = wmma32(a, b, acc);
    }
    const int col = lane & 15, rb = (lane >> 4) << 3;
#pragma unroll
    for (int i = 0; i < 8; ++i)
      Dst[(mt * 16 + rb + i) * (NTPL * DT) + tt * DT + nt * 16 + col] =
          (_Float16)acc[i];
  }
  __syncthreads();

  // ---- Phase 3: per-(pair,head) softmax over 4 templates (fp32 VALU) ------
  if (tid < TM * NH) {
    const int m = tid >> 2, h = tid & 3;
    float q[HD];
    {
      v8h q0 = *(const v8h*)(sQ + m * HID + h * HD);
      v8h q1 = *(const v8h*)(sQ + m * HID + h * HD + 8);
#pragma unroll
      for (int d = 0; d < 8; ++d) { q[d] = (float)q0[d]; q[d + 8] = (float)q1[d]; }
    }
    float lg[NTPL];
#pragma unroll
    for (int tt = 0; tt < NTPL; ++tt) {
      const _Float16* kp = sK + m * (NTPL * DT) + tt * DT + h * HD;
      v8h k0 = *(const v8h*)(kp);
      v8h k1 = *(const v8h*)(kp + 8);
      float s = sBias[tt];
#pragma unroll
      for (int d = 0; d < 8; ++d)
        s += q[d] * (float)k0[d] + q[d + 8] * (float)k1[d];
      lg[tt] = s;
    }
    float mx = lg[0];
#pragma unroll
    for (int tt = 1; tt < NTPL; ++tt) mx = fmaxf(mx, lg[tt]);
    float sum = 0.0f;
#pragma unroll
    for (int tt = 0; tt < NTPL; ++tt) { lg[tt] = __expf(lg[tt] - mx); sum += lg[tt]; }
    const float inv = 1.0f / sum;
    float o[HD];
#pragma unroll
    for (int d = 0; d < HD; ++d) o[d] = 0.0f;
#pragma unroll
    for (int tt = 0; tt < NTPL; ++tt) {
      const float a = lg[tt] * inv;
      const _Float16* vp = sV + m * (NTPL * DT) + tt * DT + h * HD;
      v8h v0 = *(const v8h*)(vp);
      v8h v1 = *(const v8h*)(vp + 8);
#pragma unroll
      for (int d = 0; d < 8; ++d) {
        o[d]     += a * (float)v0[d];
        o[d + 8] += a * (float)v1[d];
      }
    }
    v8h o0, o1;
#pragma unroll
    for (int d = 0; d < 8; ++d) { o0[d] = (_Float16)o[d]; o1[d] = (_Float16)o[d + 8]; }
    *(v8h*)(sO + m * HID + h * HD)     = o0;
    *(v8h*)(sO + m * HID + h * HD + 8) = o1;
  }
  __syncthreads();

  // ---- Phase 4: out = o @ wo + bo  [TM x DZ], K=64 ------------------------
#pragma unroll
  for (int job = wave; job < 16; job += 8) {
    const int mt = job >> 3, nt = job & 7;
    v8f acc = {};
#pragma unroll
    for (int k0 = 0; k0 < HID; k0 += 32) {
      v16h a = load_a_frag(sO + (mt * 16) * HID + k0, HID);
      v16h b = load_b_frag(sWoT + (nt * 16) * HID + k0, HID);
      acc = wmma32(a, b, acc);
    }
    const int col = lane & 15, rb = (lane >> 4) << 3;
#pragma unroll
    for (int i = 0; i < 8; ++i) {
      const int row = mt * 16 + rb + i;
      const int c   = nt * 16 + col;
      gOut[(size_t)(p0 + row) * DZ + c] = acc[i] + sBo[c];
    }
  }
}

extern "C" void kernel_launch(void* const* d_in, const int* in_sizes, int n_in,
                              void* d_out, int out_size, void* d_ws, size_t ws_size,
                              hipStream_t stream) {
  (void)in_sizes; (void)n_in; (void)out_size; (void)ws_size;
  const float* t    = (const float*)d_in[0];
  const float* z    = (const float*)d_in[1];
  const float* mask = (const float*)d_in[2];
  const float* wq   = (const float*)d_in[3];
  const float* wk   = (const float*)d_in[4];
  const float* wv   = (const float*)d_in[5];
  const float* wo   = (const float*)d_in[6];
  const float* bo   = (const float*)d_in[7];
  float*     out  = (float*)d_out;
  _Float16*  wpre = (_Float16*)d_ws;   // 48 KB used

  // 1) one-shot weight transpose/convert into workspace
  prep_weights<<<dim3(WPRE_TOTAL / 256), dim3(256), 0, stream>>>(wq, wk, wv, wo, wpre);

  // 2) fused attention kernel (TDM pulls weights from workspace into LDS)
  dim3 grid(NPAIR / TM);   // 4608 blocks
  dim3 block(256);         // 8 wave32
  tpa_kernel<<<grid, block, 0, stream>>>(t, z, mask, wpre, bo, out);
}
